// LSTMModel_69080253989482
// MI455X (gfx1250) — matrix operand order repaired and checked by reading, not compiled, not run on hardware
//
#include <hip/hip_runtime.h>

// ---------------------------------------------------------------------------
// LSTM forward on MI455X (gfx1250, wave32, WMMA).
// B=128, T=1024, D=256, H=512, O=128.
// ---------------------------------------------------------------------------

typedef __attribute__((ext_vector_type(16))) _Float16 v16h;
typedef __attribute__((ext_vector_type(8)))  _Float16 v8h;
typedef __attribute__((ext_vector_type(8)))  float    v8f;

#define TSTEPS 1024
#define BATCH  128
#define DIN    256
#define HID    512
#define GATES  2048   // 4*HID

// LDS row strides in halves, padded to dodge bank conflicts (rows stay 16B aligned)
#define WH_STRIDE 520   // 512 + 8
#define WX_STRIDE 264   // 256 + 8
#define LDS_BYTES (64 * WH_STRIDE * 2 + 64 * WX_STRIDE * 2)   // 100,352 B < 320 KB

#if __has_builtin(__builtin_amdgcn_global_load_async_to_lds_b128)
#define HAVE_ASYNC_COPY 1
#else
#define HAVE_ASYNC_COPY 0
#endif

#if HAVE_ASYNC_COPY
// The builtin takes pointers to 16-byte int vectors (per the clang signature:
// "vV4i*1V4i*3IiIi" — global int4* source, LDS int4* destination, imm, imm).
typedef int v4i_vs __attribute__((vector_size(16)));
typedef __attribute__((address_space(1))) v4i_vs as1_v4i;
typedef __attribute__((address_space(3))) v4i_vs as3_v4i;

// 16B per lane, global -> LDS, tracked by ASYNCcnt (no VGPR data movement).
static __device__ __forceinline__ void async_copy16(const _Float16* gsrc,
                                                    _Float16* ldst) {
  // flat global address == AS1 address numerically
  as1_v4i* g = (as1_v4i*)(unsigned long long)gsrc;
  // LDS aperture: flat addr[31:0] is the LDS byte offset (ISA 10.2).
  as3_v4i* l = (as3_v4i*)(unsigned int)(unsigned long long)ldst;
  __builtin_amdgcn_global_load_async_to_lds_b128(g, l, 0, 0);
}
static __device__ __forceinline__ void async_wait0() {
#if __has_builtin(__builtin_amdgcn_s_wait_asynccnt)
  __builtin_amdgcn_s_wait_asynccnt(0);
#else
  asm volatile("s_wait_asynccnt 0x0" ::: "memory");
#endif
}
#endif

static __device__ __forceinline__ v8f wmma_f16(v16h a, v16h b, v8f c) {
  // (neg_a, A, neg_b, B, c_mod, C, reuse_a, reuse_b)
  return __builtin_amdgcn_wmma_f32_16x16x32_f16(false, a, false, b, (short)0, c,
                                                false, false);
}

// A operand (16x32 f16): lane holds row M = lane%16.
// Layout: lanes 0-15 hold K = 0..15; lanes 16-31 hold K = 16..31 (matches B).
static __device__ __forceinline__ v16h load_a_tile(const _Float16* base, int hi) {
  v8h lo = *(const v8h*)(base + 16 * hi);
  v8h hh = *(const v8h*)(base + 16 * hi + 8);
  return __builtin_shufflevector(lo, hh, 0, 1, 2, 3, 4, 5, 6, 7,
                                 8, 9, 10, 11, 12, 13, 14, 15);
}

// B operand (32x16 f16): lane holds column N = lane%16; lanes 0-15 K=0..15,
// lanes 16-31 K=16..31 (caller adds 16*hi). Two 16B loads (rows only 16B aligned).
static __device__ __forceinline__ v16h load_b_tile(const _Float16* p) {
  v8h lo = *(const v8h*)p;
  v8h hh = *(const v8h*)(p + 8);
  return __builtin_shufflevector(lo, hh, 0, 1, 2, 3, 4, 5, 6, 7,
                                 8, 9, 10, 11, 12, 13, 14, 15);
}

static __device__ __forceinline__ float fast_sigmoid(float x) {
  return __builtin_amdgcn_rcpf(1.0f + __expf(-x));
}
static __device__ __forceinline__ float fast_tanh(float x) {
  return 2.0f * fast_sigmoid(2.0f * x) - 1.0f;
}

// ---------------------------------------------------------------------------
// Setup kernels
// ---------------------------------------------------------------------------

__global__ void k_cvt_f16(const float* __restrict__ s, _Float16* __restrict__ d,
                          long n) {
  long i = (long)blockIdx.x * blockDim.x + threadIdx.x;
  long idx = i * 8;
  if (idx + 8 <= n) {
    float4 a = ((const float4*)s)[i * 2 + 0];
    float4 b = ((const float4*)s)[i * 2 + 1];
    v8h o = {(_Float16)a.x, (_Float16)a.y, (_Float16)a.z, (_Float16)a.w,
             (_Float16)b.x, (_Float16)b.y, (_Float16)b.z, (_Float16)b.w};
    *(v8h*)(d + idx) = o;
  }
}

__global__ void k_prep(const float* __restrict__ bx, const float* __restrict__ bh,
                       float* __restrict__ bxh, _Float16* __restrict__ hbuf,
                       unsigned int* __restrict__ sync) {
  int idx = blockIdx.x * blockDim.x + threadIdx.x;  // 512*256 = 131072 threads
  if (idx < 2 * BATCH * HID) hbuf[idx] = (_Float16)0.0f;  // zero both h buffers
  if (idx < GATES) bxh[idx] = bx[idx] + bh[idx];
  if (idx < 64) sync[idx] = 0u;
}

// ---------------------------------------------------------------------------
// Persistent LSTM recurrence.
// 32 blocks (one H column-tile each) x 128 threads (4 waves).
// Wave w owns TWO batch row-tiles (w and w+4), so every LDS B-tile read feeds
// two WMMAs: halves DS traffic per unit of math vs the 1-row-tile version.
// ---------------------------------------------------------------------------

__global__ __launch_bounds__(128, 1) void k_lstm(
    const _Float16* __restrict__ x16,   // [B, T, D] f16
    const _Float16* __restrict__ Wx16,  // [4H, D]   f16
    const _Float16* __restrict__ Wh16,  // [4H, H]   f16
    const float* __restrict__ bxh,      // [4H]
    _Float16* __restrict__ hbuf,        // [2, B, H] f16, double buffered
    float* __restrict__ hT_out,         // [B, H] f32  (d_out part 1)
    unsigned int* __restrict__ sync) {
  extern __shared__ char smem[];
  _Float16* ldsWh = (_Float16*)smem;                         // [64][WH_STRIDE]
  _Float16* ldsWx = (_Float16*)(smem + 64 * WH_STRIDE * 2);  // [64][WX_STRIDE]

  const int ct   = blockIdx.x;           // H column-tile, 0..31
  const int wave = threadIdx.x >> 5;     // 0..3 -> row tiles (wave, wave+4)
  const int lane = threadIdx.x & 31;
  const int n    = lane & 15;            // tile column (B/C/D) or tile row (A)
  const int hi   = lane >> 4;            // K-half selector

  // ---- one-time preload of this block's Wh/Wx column slices into LDS ----
  for (int s = wave; s < 64; s += 4) {
    const int q = s >> 4, j = s & 15;               // gate, column-in-tile
    const int grow = q * HID + ct * 16 + j;         // global gate row
    const _Float16* srcWh = Wh16 + (size_t)grow * HID;
    _Float16* dstWh = ldsWh + s * WH_STRIDE;
    const _Float16* srcWx = Wx16 + (size_t)grow * DIN;
    _Float16* dstWx = ldsWx + s * WX_STRIDE;
#if HAVE_ASYNC_COPY
    for (int kc = lane; kc < HID / 8; kc += 32)
      async_copy16(srcWh + kc * 8, dstWh + kc * 8);
    for (int kc = lane; kc < DIN / 8; kc += 32)
      async_copy16(srcWx + kc * 8, dstWx + kc * 8);
#else
    for (int kc = lane; kc < HID / 8; kc += 32)
      ((v8h*)dstWh)[kc] = ((const v8h*)srcWh)[kc];
    for (int kc = lane; kc < DIN / 8; kc += 32)
      ((v8h*)dstWx)[kc] = ((const v8h*)srcWx)[kc];
#endif
  }
#if HAVE_ASYNC_COPY
  async_wait0();
#endif
  __syncthreads();

  // Bias (per gate, depends only on this lane's column) — constant over t.
  float bias[4];
#pragma unroll
  for (int q = 0; q < 4; ++q) bias[q] = bxh[q * HID + ct * 16 + n];

  const int row0 = wave * 16 + n;        // batch row for A operand, tile 0
  const size_t ROWD = (size_t)64 * TSTEPS * DIN;   // +64 batch rows in x16
  const _Float16* xrow0 = x16 + (size_t)row0 * TSTEPS * DIN;

  v8f creg[2];  // cell-state tiles live in registers for all 1024 steps
#pragma unroll
  for (int p = 0; p < 2; ++p)
#pragma unroll
    for (int r = 0; r < 8; ++r) creg[p][r] = 0.0f;

  for (int t = 0; t < TSTEPS; ++t) {
    const _Float16* hprev = hbuf + (size_t)(t & 1) * (BATCH * HID);
    _Float16* hnext = hbuf + (size_t)((t + 1) & 1) * (BATCH * HID);

    v8f acc[2][4];
#pragma unroll
    for (int p = 0; p < 2; ++p)
#pragma unroll
      for (int q = 0; q < 4; ++q)
#pragma unroll
        for (int r = 0; r < 8; ++r) acc[p][q][r] = bias[q];

    // ---- x_t projection: K = 256 against Wx slice in LDS ----
    const _Float16* xa0 = xrow0 + (size_t)t * DIN;
    const _Float16* xa1 = xa0 + ROWD;
#pragma unroll 2
    for (int k = 0; k < DIN; k += 32) {
      v16h A0 = load_a_tile(xa0 + k, hi);
      v16h A1 = load_a_tile(xa1 + k, hi);
#pragma unroll
      for (int q = 0; q < 4; ++q) {
        v16h Bv = load_b_tile(ldsWx + (q * 16 + n) * WX_STRIDE + k + 16 * hi);
        acc[0][q] = wmma_f16(A0, Bv, acc[0][q]);
        acc[1][q] = wmma_f16(A1, Bv, acc[1][q]);
      }
    }

    // ---- recurrent projection: K = 512 against Wh slice in LDS ----
    const _Float16* ha0 = hprev + (size_t)row0 * HID;
    const _Float16* ha1 = ha0 + (size_t)64 * HID;
#pragma unroll 2
    for (int k = 0; k < HID; k += 32) {
      v16h A0 = load_a_tile(ha0 + k, hi);
      v16h A1 = load_a_tile(ha1 + k, hi);
#pragma unroll
      for (int q = 0; q < 4; ++q) {
        v16h Bv = load_b_tile(ldsWh + (q * 16 + n) * WH_STRIDE + k + 16 * hi);
        acc[0][q] = wmma_f16(A0, Bv, acc[0][q]);
        acc[1][q] = wmma_f16(A1, Bv, acc[1][q]);
      }
    }

    // ---- elementwise cell update (C/D layout: VGPR r -> row r + 8*hi) ----
    v8f hreg[2];
#pragma unroll
    for (int p = 0; p < 2; ++p) {
#pragma unroll
      for (int r = 0; r < 8; ++r) {
        float iv = fast_sigmoid(acc[p][0][r]);
        float fv = fast_sigmoid(acc[p][1][r]);
        float gv = fast_tanh(acc[p][2][r]);
        float ov = fast_sigmoid(acc[p][3][r]);
        float cv = creg[p][r] * fv + iv * gv;
        creg[p][r] = cv;
        hreg[p][r] = ov * fast_tanh(cv);
      }
    }

    // ---- publish h tiles (f16); 16 lanes/row => coalesced b16 stores ----
    _Float16* hb0 = hnext + (size_t)(wave * 16) * HID + ct * 16 + n;
    _Float16* hb1 = hb0 + (size_t)64 * HID;
#pragma unroll
    for (int r = 0; r < 8; ++r) {
      hb0[(size_t)(r + 8 * hi) * HID] = (_Float16)hreg[0][r];
      hb1[(size_t)(r + 8 * hi) * HID] = (_Float16)hreg[1][r];
    }

    if (t == TSTEPS - 1) {
      float* ho0 = hT_out + (size_t)(wave * 16) * HID + ct * 16 + n;
      float* ho1 = ho0 + (size_t)64 * HID;
#pragma unroll
      for (int r = 0; r < 8; ++r) {
        ho0[(size_t)(r + 8 * hi) * HID] = hreg[0][r];
        ho1[(size_t)(r + 8 * hi) * HID] = hreg[1][r];
      }
    }

    // ---- device-wide step barrier (monotonic counter, no reset races) ----
    __threadfence();   // release my h stores device-wide
    __syncthreads();   // whole block done with this step
    if (threadIdx.x == 0) {
      __hip_atomic_fetch_add(sync, 1u, __ATOMIC_RELEASE, __HIP_MEMORY_SCOPE_AGENT);
      const unsigned tgt = (unsigned)(t + 1) * gridDim.x;
      while (__hip_atomic_load(sync, __ATOMIC_ACQUIRE, __HIP_MEMORY_SCOPE_AGENT) < tgt)
        __builtin_amdgcn_s_sleep(1);
    }
    __syncthreads();
  }
}

// ---------------------------------------------------------------------------
// Final FC: out = hT @ Wfc.T + bfc   ([128,512] x [512,128])
// ---------------------------------------------------------------------------

__global__ __launch_bounds__(256) void k_fc(const _Float16* __restrict__ h16,
                                            const _Float16* __restrict__ Wfc16,
                                            const float* __restrict__ bfc,
                                            float* __restrict__ out) {
  const int ct   = blockIdx.x;        // O column-tile, 0..7
  const int wave = threadIdx.x >> 5;  // B row-tile, 0..7
  const int lane = threadIdx.x & 31;
  const int n = lane & 15, hi = lane >> 4;

  const float bias = bfc[ct * 16 + n];
  v8f acc;
#pragma unroll
  for (int r = 0; r < 8; ++r) acc[r] = bias;

  const _Float16* arow = h16 + (size_t)(wave * 16 + n) * HID;
  const _Float16* brow = Wfc16 + (size_t)(ct * 16 + n) * HID;  // 1KB aligned rows
#pragma unroll 4
  for (int k = 0; k < HID; k += 32) {
    v16h A = load_a_tile(arow + k, hi);
    v16h Bv = *(const v16h*)(brow + k + 16 * hi);
    acc = wmma_f16(A, Bv, acc);
  }
  float* ob = out + (size_t)(wave * 16) * 128 + ct * 16 + n;
#pragma unroll
  for (int r = 0; r < 8; ++r) ob[(size_t)(r + 8 * hi) * 128] = acc[r];
}

// ---------------------------------------------------------------------------
// Host launcher
// ---------------------------------------------------------------------------

extern "C" void kernel_launch(void* const* d_in, const int* in_sizes, int n_in,
                              void* d_out, int out_size, void* d_ws,
                              size_t ws_size, hipStream_t stream) {
  (void)in_sizes; (void)n_in; (void)out_size; (void)ws_size;

  const float* x   = (const float*)d_in[0];
  // d_in[1] = EC_in : dead code in reference
  const float* Wx  = (const float*)d_in[2];
  const float* bx  = (const float*)d_in[3];
  const float* Wh  = (const float*)d_in[4];
  const float* bh  = (const float*)d_in[5];
  // d_in[6], d_in[7] = Wec, bec : dead code in reference
  const float* Wfc = (const float*)d_in[8];
  const float* bfc = (const float*)d_in[9];

  char* ws = (char*)d_ws;
  size_t off = 0;
  auto carve = [&](size_t bytes) -> char* {
    char* p = ws + off;
    off += (bytes + 255) & ~(size_t)255;
    return p;
  };

  const long NX  = (long)BATCH * TSTEPS * DIN;  // 33,554,432
  const long NWX = (long)GATES * DIN;           //    524,288
  const long NWH = (long)GATES * HID;           //  1,048,576
  const long NWF = (long)128 * HID;             //     65,536

  _Float16*     x16   = (_Float16*)carve((size_t)NX * 2);
  _Float16*     Wx16  = (_Float16*)carve((size_t)NWX * 2);
  _Float16*     Wh16  = (_Float16*)carve((size_t)NWH * 2);
  _Float16*     Wfc16 = (_Float16*)carve((size_t)NWF * 2);
  float*        bxh   = (float*)carve((size_t)GATES * 4);
  _Float16*     hbuf  = (_Float16*)carve((size_t)2 * BATCH * HID * 2);
  unsigned int* syncp = (unsigned int*)carve(256);

  // f32 -> f16 conversions (8 elems/thread; all counts divisible by 2048)
  k_cvt_f16<<<(int)(NX  / 2048), 256, 0, stream>>>(x,   x16,   NX);
  k_cvt_f16<<<(int)(NWX / 2048), 256, 0, stream>>>(Wx,  Wx16,  NWX);
  k_cvt_f16<<<(int)(NWH / 2048), 256, 0, stream>>>(Wh,  Wh16,  NWH);
  k_cvt_f16<<<(int)(NWF / 2048), 256, 0, stream>>>(Wfc, Wfc16, NWF);

  // combined bias, zero h buffers, zero barrier counter
  k_prep<<<512, 256, 0, stream>>>(bx, bh, bxh, hbuf, syncp);

  // persistent recurrence: 32 blocks x 4 waves, ~98KB LDS each (fits one WGP)
  k_lstm<<<32, 128, LDS_BYTES, stream>>>(x16, Wx16, Wh16, bxh, hbuf,
                                         (float*)d_out, syncp);

  // final projection; after 1024 steps h lives in buffer 0
  k_fc<<<8, 256, 0, stream>>>(hbuf, Wfc16, bfc, (float*)d_out + BATCH * HID);
}
